// GNNBase_27023934226710
// MI455X (gfx1250) — matrix operand order repaired and checked by reading, not compiled, not run on hardware
//
#include <hip/hip_runtime.h>
#include <math.h>

// ---------------------------------------------------------------------------
// GNN forward (EmbedConv + 2x TransformerConv) for MI455X / gfx1250.
// GEMMs: v_wmma_f32_16x16x32_bf16, packed-fragment weights, async A staging.
// LN / softmax / scatter in f32 with global atomics.
// ---------------------------------------------------------------------------

typedef __attribute__((ext_vector_type(16))) __bf16 v16bf;
typedef __attribute__((ext_vector_type(8)))  float  v8f;
typedef __attribute__((ext_vector_type(4)))  int    v4i;

#define NN  32768            // nodes
#define EE  (NN * 8)         // edges

// ---- bf16 helpers (memory interface kept as unsigned short for host safety) ----
__device__ __forceinline__ unsigned short f2bf(float f) {
  unsigned int u = __float_as_uint(f);
  u = u + 0x7FFFu + ((u >> 16) & 1u);   // round-to-nearest-even
  return (unsigned short)(u >> 16);
}
__device__ __forceinline__ __bf16 bits2bf(unsigned short s) {
  union { unsigned short u; __bf16 b; } x; x.u = s; return x.b;
}

// ---- async global->LDS staging (gfx1250), guarded ----
__device__ __forceinline__ void wait_async0() {
#if __has_builtin(__builtin_amdgcn_s_wait_asynccnt)
  __builtin_amdgcn_s_wait_asynccnt(0);
#else
  asm volatile("s_wait_asynccnt 0" ::: "memory");
#endif
}

// ---- WMMA fragment builders (CDNA5 ISA 7.12.2 layouts) ----
// A (16x32): lane = half*16 + m ; VGPR v holds K = 2v + 8*half (+8 if v>=4)
__device__ __forceinline__ v16bf load_a_frag_u16(const unsigned short* sm, int ld, int k0) {
  int lane = threadIdx.x & 31;
  int half = lane >> 4, m = lane & 15;
  v16bf a;
#pragma unroll
  for (int v = 0; v < 8; ++v) {
    int kb = 2 * v + 8 * half + (v >= 4 ? 8 : 0);
    a[2 * v]     = bits2bf(sm[m * ld + k0 + kb]);
    a[2 * v + 1] = bits2bf(sm[m * ld + k0 + kb + 1]);
  }
  return a;
}
// A fragment from an f32 16x32 LDS tile (convert at build time)
__device__ __forceinline__ v16bf make_a_frag_f32(const float* sm) {
  int lane = threadIdx.x & 31;
  int half = lane >> 4, m = lane & 15;
  v16bf a;
#pragma unroll
  for (int v = 0; v < 8; ++v) {
    int kb = 2 * v + 8 * half + (v >= 4 ? 8 : 0);
    a[2 * v]     = bits2bf(f2bf(sm[m * 32 + kb]));
    a[2 * v + 1] = bits2bf(f2bf(sm[m * 32 + kb + 1]));
  }
  return a;
}
// B fragment from packed weights: per (kb,nb) tile each lane's 16 bf16 are
// contiguous 32B -> two global_load_b128.
__device__ __forceinline__ v16bf load_bp_frag(const unsigned short* P, int nblocks,
                                              int kb, int nb) {
  int lane = threadIdx.x & 31;
  const unsigned short* base = P + ((size_t)(kb * nblocks + nb) * 32 + lane) * 16;
  return *(const v16bf*)base;
}
__device__ __forceinline__ v8f wmma_bf16(v16bf a, v16bf b, v8f c) {
  return __builtin_amdgcn_wmma_f32_16x16x32_bf16(false, a, false, b, (short)0, c,
                                                 false, false);
}

// ---- float atomic max via CAS ----
__device__ __forceinline__ void atomicMaxF(float* addr, float val) {
  unsigned int* ai = (unsigned int*)addr;
  unsigned int old = *ai;
  while (true) {
    float f = __uint_as_float(old);
    if (f >= val) break;
    unsigned int assumed = old;
    old = atomicCAS(ai, assumed, __float_as_uint(val));
    if (old == assumed) break;
  }
}

// ===========================================================================
// Weight conversion+packing: f32 [K,N] -> bf16 fragment-packed [Kpad/32][N/16]
// tiles; within a tile: [lane 0..31][j 0..15], element (k,n) =
// (32*kb + 8*(lane>>4) + (j<8 ? j : j+8), 16*nb + (lane&15)).  Zero-pad k >= K.
// ===========================================================================
__global__ void cvt_pack_kernel(const float* __restrict__ src, unsigned short* __restrict__ dst,
                                int K, int Nld, int Kpad) {
  int i = blockIdx.x * 256 + threadIdx.x;
  if (i >= Kpad * Nld) return;
  int j = i & 15;
  int l = (i >> 4) & 31;
  int rest = i >> 9;                 // kb*nblocks + nb
  int nblocks = Nld >> 4;
  int nb = rest % nblocks, kb = rest / nblocks;
  int k = 32 * kb + 8 * (l >> 4) + (j < 8 ? j : j + 8);
  int col = 16 * nb + (l & 15);
  dst[i] = (k < K) ? f2bf(src[k * Nld + col]) : (unsigned short)0;
}

__global__ void fill_kernel(float* __restrict__ p, float v, int n) {
  int i = blockIdx.x * 256 + threadIdx.x;
  if (i < n) p[i] = v;
}

// ===========================================================================
// EmbedConv: per-edge MLP (24->128 relu LN, 128->128 relu LN), scatter-add.
// One block = 16 edges, 8 waves cover the 128 output columns.
// ===========================================================================
__global__ void embed_kernel(const float* __restrict__ x, const float* __restrict__ ea,
                             const int* __restrict__ ei,
                             const unsigned short* __restrict__ W1p, const float* __restrict__ b1,
                             const float* __restrict__ g1, const float* __restrict__ be1,
                             const unsigned short* __restrict__ W2p, const float* __restrict__ b2,
                             const float* __restrict__ g2, const float* __restrict__ be2,
                             float* __restrict__ h0) {
  __shared__ unsigned short sIn[16 * 32];   // input tile, bf16, K padded to 32
  __shared__ float          sT[16 * 128];   // f32 intermediate
  __shared__ unsigned short sA[16 * 128];   // bf16 A for GEMM2
  __shared__ float sMean[16], sRstd[16];

  const int e0 = blockIdx.x * 16;
  for (int idx = threadIdx.x; idx < 512; idx += 256) {   // cat(x[src], edge_attr)
    int r = idx >> 5, c = idx & 31, e = e0 + r;
    int s = ei[e];
    float v = 0.f;
    if (c < 16)      v = x[s * 16 + c];
    else if (c < 24) v = ea[e * 8 + (c - 16)];
    sIn[idx] = f2bf(v);
  }
  __syncthreads();

  const int wave = threadIdx.x >> 5;
  const int lane = threadIdx.x & 31;
  const int half = lane >> 4, n = lane & 15;
  const int col = wave * 16 + n;

  { // GEMM1: [16,32] x [32,128] -> relu into sT
    v8f cz = {};
    v16bf a = load_a_frag_u16(sIn, 32, 0);
    v16bf b = load_bp_frag(W1p, 8, 0, wave);
    v8f c = wmma_bf16(a, b, cz);
    float bn = b1[col];
#pragma unroll
    for (int r = 0; r < 8; ++r) {
      float v = c[r] + bn;
      sT[(r + 8 * half) * 128 + col] = v > 0.f ? v : 0.f;
    }
  }
  __syncthreads();
  if (threadIdx.x < 16) { // LN row stats
    float m = 0.f;
    for (int c = 0; c < 128; ++c) m += sT[threadIdx.x * 128 + c];
    m *= (1.f / 128.f);
    float var = 0.f;
    for (int c = 0; c < 128; ++c) { float d = sT[threadIdx.x * 128 + c] - m; var += d * d; }
    sMean[threadIdx.x] = m;
    sRstd[threadIdx.x] = rsqrtf(var * (1.f / 128.f) + 1e-5f);
  }
  __syncthreads();
  for (int idx = threadIdx.x; idx < 2048; idx += 256) { // normalize -> bf16 A
    int r = idx >> 7, c = idx & 127;
    float v = (sT[idx] - sMean[r]) * sRstd[r] * g1[c] + be1[c];
    sA[idx] = f2bf(v);
  }
  __syncthreads();

  { // GEMM2: [16,128] x [128,128] -> relu into sT
    v8f c = {};
#pragma unroll
    for (int k0 = 0; k0 < 128; k0 += 32) {
      v16bf a = load_a_frag_u16(sA, 128, k0);
      v16bf b = load_bp_frag(W2p, 8, k0 >> 5, wave);
      c = wmma_bf16(a, b, c);
    }
    float bn = b2[col];
#pragma unroll
    for (int r = 0; r < 8; ++r) {
      float v = c[r] + bn;
      sT[(r + 8 * half) * 128 + col] = v > 0.f ? v : 0.f;
    }
  }
  __syncthreads();
  if (threadIdx.x < 16) { // LN row stats (2nd)
    float m = 0.f;
    for (int c = 0; c < 128; ++c) m += sT[threadIdx.x * 128 + c];
    m *= (1.f / 128.f);
    float var = 0.f;
    for (int c = 0; c < 128; ++c) { float d = sT[threadIdx.x * 128 + c] - m; var += d * d; }
    sMean[threadIdx.x] = m;
    sRstd[threadIdx.x] = rsqrtf(var * (1.f / 128.f) + 1e-5f);
  }
  __syncthreads();
  for (int idx = threadIdx.x; idx < 2048; idx += 256) { // normalize + scatter-add
    int r = idx >> 7, c = idx & 127;
    float v = (sT[idx] - sMean[r]) * sRstd[r] * g2[c] + be2[c];
    int d = ei[EE + e0 + r];
    atomicAdd(&h0[d * 128 + c], v);
  }
}

// ===========================================================================
// Node GEMM: D[M,256] = A[M,K](f32) x Bpacked[Kpad,256](bf16) + bias.
// Block = 128 threads (4 waves); wave w owns N-tiles 4w..4w+3, so one block
// produces a full 16x256 row-block.  One A fragment feeds 4 WMMAs per K-step.
// A tile staged to LDS with async global->LDS (f32), converted at frag build.
// ===========================================================================
__global__ void gemm_bias_kernel(const float* __restrict__ A, const unsigned short* __restrict__ Bp,
                                 const float* __restrict__ bias, float* __restrict__ D,
                                 int K) {
  __shared__ float sAf[16 * 32];
  const int m0 = blockIdx.x * 16;
  const int wave = threadIdx.x >> 5;
  const int lane = threadIdx.x & 31;
  v8f acc[4] = {};
  for (int k0 = 0; k0 < K; k0 += 32) {
#if __has_builtin(__builtin_amdgcn_global_load_async_to_lds_b128)
    {
      int r = threadIdx.x >> 3, c4 = (threadIdx.x & 7) * 4;   // 128 thr x 16B = 2KB tile
      const float* gsrc = A + (size_t)(m0 + r) * K + k0 + c4;
      __builtin_amdgcn_global_load_async_to_lds_b128(
          (v4i*)(void*)gsrc, (v4i*)(void*)&sAf[r * 32 + c4], 0, 0);
    }
    wait_async0();
    __syncthreads();
#else
    for (int idx = threadIdx.x; idx < 512; idx += 128) {
      int r = idx >> 5, c = idx & 31;
      sAf[idx] = A[(size_t)(m0 + r) * K + k0 + c];
    }
    __syncthreads();
#endif
    v16bf a = make_a_frag_f32(sAf);
    int kb = k0 >> 5;
#pragma unroll
    for (int t = 0; t < 4; ++t) {
      v16bf b = load_bp_frag(Bp, 16, kb, wave * 4 + t);
      acc[t] = wmma_bf16(a, b, acc[t]);
    }
    __syncthreads();
  }
  const int half = lane >> 4, n = lane & 15;
#pragma unroll
  for (int t = 0; t < 4; ++t) {
    int n0 = (wave * 4 + t) * 16;
    float bn = bias[n0 + n];
#pragma unroll
    for (int r = 0; r < 8; ++r)
      D[(size_t)(m0 + r + 8 * half) * 256 + n0 + n] = acc[t][r] + bn;
  }
}

// ===========================================================================
// Attention: score / exp+sum / scatter / finalize.  e = edge_attr @ We is
// recomputed in registers (8xFMA per channel) instead of a 256MB buffer.
// ===========================================================================
__global__ void score_kernel(const float* __restrict__ q, const float* __restrict__ k,
                             const float* __restrict__ ea, const float* __restrict__ We,
                             const int* __restrict__ ei, float* __restrict__ s,
                             float* __restrict__ mbuf) {
  int idx = blockIdx.x * 256 + threadIdx.x;
  if (idx >= EE * 4) return;
  int e = idx >> 2, h = idx & 3;
  int src = ei[e], dst = ei[EE + e];
  float a8[8];
#pragma unroll
  for (int j = 0; j < 8; ++j) a8[j] = ea[e * 8 + j];
  const float* qr = q + dst * 256 + h * 64;
  const float* kr = k + src * 256 + h * 64;
  const float* Wc = We + h * 64;
  float acc = 0.f;
  for (int c = 0; c < 64; ++c) {
    float ec = 0.f;
#pragma unroll
    for (int j = 0; j < 8; ++j) ec += a8[j] * Wc[j * 256 + c];
    acc += qr[c] * (kr[c] + ec);
  }
  acc *= 0.125f;                     // 1/sqrt(64)
  s[idx] = acc;
  atomicMaxF(&mbuf[dst * 4 + h], acc);
}

__global__ void expsum_kernel(const int* __restrict__ ei, float* __restrict__ s,
                              const float* __restrict__ mbuf, float* __restrict__ z) {
  int idx = blockIdx.x * 256 + threadIdx.x;
  if (idx >= EE * 4) return;
  int e = idx >> 2, h = idx & 3;
  int dst = ei[EE + e];
  float p = __expf(s[idx] - mbuf[dst * 4 + h]);
  s[idx] = p;
  atomicAdd(&z[dst * 4 + h], p);
}

__global__ void scatter_kernel(const float* __restrict__ v, const float* __restrict__ ea,
                               const float* __restrict__ We, const int* __restrict__ ei,
                               const float* __restrict__ s, const float* __restrict__ z,
                               float* __restrict__ attn) {
  int idx = blockIdx.x * 256 + threadIdx.x;
  if (idx >= EE * 4) return;
  int e = idx >> 2, h = idx & 3;
  int src = ei[e], dst = ei[EE + e];
  float w = s[idx] / z[dst * 4 + h];
  float a8[8];
#pragma unroll
  for (int j = 0; j < 8; ++j) a8[j] = ea[e * 8 + j];
  const float* vr = v + src * 256 + h * 64;
  const float* Wc = We + h * 64;
  float* outr = attn + dst * 256 + h * 64;
  for (int c = 0; c < 64; ++c) {
    float ec = 0.f;
#pragma unroll
    for (int j = 0; j < 8; ++j) ec += a8[j] * Wc[j * 256 + c];
    atomicAdd(&outr[c], w * (vr[c] + ec));
  }
}

__global__ void finalize_kernel(const float* __restrict__ attn, const float* __restrict__ sk,
                                float* __restrict__ out, int n) {
  int i = blockIdx.x * 256 + threadIdx.x;
  if (i < n) { float v = attn[i] + sk[i]; out[i] = v > 0.f ? v : 0.f; }
}

// ===========================================================================
// Host driver
// ===========================================================================
extern "C" void kernel_launch(void* const* d_in, const int* in_sizes, int n_in,
                              void* d_out, int out_size, void* d_ws, size_t ws_size,
                              hipStream_t stream) {
  (void)in_sizes; (void)n_in; (void)out_size; (void)ws_size;
  const float* x     = (const float*)d_in[0];
  const int*   ei    = (const int*)  d_in[1];
  const float* ea    = (const float*)d_in[2];
  const float* e_W1  = (const float*)d_in[3];
  const float* e_b1  = (const float*)d_in[4];
  const float* e_g1  = (const float*)d_in[5];
  const float* e_be1 = (const float*)d_in[6];
  const float* e_W2  = (const float*)d_in[7];
  const float* e_b2  = (const float*)d_in[8];
  const float* e_g2  = (const float*)d_in[9];
  const float* e_be2 = (const float*)d_in[10];
  // l0: 11..19, l1: 20..28  (Wq,bq,Wk,bk,Wv,bv,We,Ws,bs)
  const float* lW[2][4]; const float* lb[2][4]; const float* lWe[2];
  for (int l = 0; l < 2; ++l) {
    int base = 11 + l * 9;
    lW[l][0] = (const float*)d_in[base + 0]; lb[l][0] = (const float*)d_in[base + 1]; // q
    lW[l][1] = (const float*)d_in[base + 2]; lb[l][1] = (const float*)d_in[base + 3]; // k
    lW[l][2] = (const float*)d_in[base + 4]; lb[l][2] = (const float*)d_in[base + 5]; // v
    lWe[l]   = (const float*)d_in[base + 6];
    lW[l][3] = (const float*)d_in[base + 7]; lb[l][3] = (const float*)d_in[base + 8]; // skip
  }

  // ---- workspace carve-up ----
  char* p = (char*)d_ws;
  auto alloc = [&](size_t bytes) -> void* {
    void* r = (void*)p; p += (bytes + 255) & ~(size_t)255; return r;
  };
  unsigned short* W1p = (unsigned short*)alloc(32 * 128 * 2);
  unsigned short* W2p = (unsigned short*)alloc(128 * 128 * 2);
  unsigned short* lWp[2][4];
  for (int l = 0; l < 2; ++l) {
    int din = (l == 0) ? 128 : 256;
    for (int w = 0; w < 4; ++w) lWp[l][w] = (unsigned short*)alloc((size_t)din * 256 * 2);
  }
  float* h0   = (float*)alloc((size_t)NN * 128 * 4);
  float* qb   = (float*)alloc((size_t)NN * 256 * 4);
  float* kb   = (float*)alloc((size_t)NN * 256 * 4);
  float* vb   = (float*)alloc((size_t)NN * 256 * 4);
  float* skb  = (float*)alloc((size_t)NN * 256 * 4);
  float* h1   = (float*)alloc((size_t)NN * 256 * 4);
  float* attn = (float*)alloc((size_t)NN * 256 * 4);
  float* sbuf = (float*)alloc((size_t)EE * 4 * 4);
  float* mbuf = (float*)alloc((size_t)NN * 4 * 4);
  float* zbuf = (float*)alloc((size_t)NN * 4 * 4);

  // ---- weight conversion + fragment packing ----
  cvt_pack_kernel<<<(32 * 128) / 256, 256, 0, stream>>>(e_W1, W1p, 24, 128, 32);
  cvt_pack_kernel<<<(128 * 128) / 256, 256, 0, stream>>>(e_W2, W2p, 128, 128, 128);
  for (int l = 0; l < 2; ++l) {
    int din = (l == 0) ? 128 : 256;
    for (int w = 0; w < 4; ++w)
      cvt_pack_kernel<<<(din * 256) / 256, 256, 0, stream>>>(lW[l][w], lWp[l][w], din, 256, din);
  }

  // ---- EmbedConv ----
  fill_kernel<<<(NN * 128) / 256, 256, 0, stream>>>(h0, 0.f, NN * 128);
  embed_kernel<<<EE / 16, 256, 0, stream>>>(x, ea, ei, W1p, e_b1, e_g1, e_be1,
                                            W2p, e_b2, e_g2, e_be2, h0);

  // ---- TransformerConv layers ----
  const float* hin = h0;
  for (int l = 0; l < 2; ++l) {
    int din = (l == 0) ? 128 : 256;
    gemm_bias_kernel<<<NN / 16, 128, 0, stream>>>(hin, lWp[l][0], lb[l][0], qb,  din);
    gemm_bias_kernel<<<NN / 16, 128, 0, stream>>>(hin, lWp[l][1], lb[l][1], kb,  din);
    gemm_bias_kernel<<<NN / 16, 128, 0, stream>>>(hin, lWp[l][2], lb[l][2], vb,  din);
    gemm_bias_kernel<<<NN / 16, 128, 0, stream>>>(hin, lWp[l][3], lb[l][3], skb, din);

    fill_kernel<<<(NN * 4) / 256, 256, 0, stream>>>(mbuf, -3.402823e38f, NN * 4);
    fill_kernel<<<(NN * 4) / 256, 256, 0, stream>>>(zbuf, 0.f, NN * 4);
    fill_kernel<<<(NN * 256) / 256, 256, 0, stream>>>(attn, 0.f, NN * 256);

    score_kernel<<<(EE * 4) / 256, 256, 0, stream>>>(qb, kb, ea, lWe[l], ei, sbuf, mbuf);
    expsum_kernel<<<(EE * 4) / 256, 256, 0, stream>>>(ei, sbuf, mbuf, zbuf);
    scatter_kernel<<<(EE * 4) / 256, 256, 0, stream>>>(vb, ea, lWe[l], ei, sbuf, zbuf, attn);

    float* outp = (l == 0) ? h1 : (float*)d_out;
    finalize_kernel<<<(NN * 256) / 256, 256, 0, stream>>>(attn, skb, outp, NN * 256);
    hin = h1;
  }
}
